// GeneralizedNewtonSchulz_49512382988719
// MI455X (gfx1250) — matrix operand order repaired and checked by reading, not compiled, not run on hardware
//
#include <hip/hip_runtime.h>

// ---------------------------------------------------------------------------
// Batched Newton-Schulz (Muon quintic) for 64 x 1024 x 1024 fp32 matrices.
// bf16 operands + fp32 accumulation via v_wmma_f32_16x16x32_bf16.
// Roofline: 15 batched 1024^3 GEMMs = 2.06 TFLOP vs ~11.5 GB HBM traffic;
// at ~5 PF dense bf16 and 23.3 TB/s these are ~0.4ms / ~0.5ms -> the kernel
// must overlap global traffic with WMMA, hence double-buffered LDS staging.
// ---------------------------------------------------------------------------

typedef __attribute__((ext_vector_type(16))) __bf16 v16bf;
typedef __attribute__((ext_vector_type(8)))  float  v8f;

#define MDIM 1024
#define BATCH 64
#define BM 128
#define BN 128
#define BK 32
#define KSTEPS (MDIM / BK)
#define NUM_ITER 5
#define INITIAL_SCALE 1.1f
#define LDS_PAD 4   // even pad: keeps 32-bit LDS fragment reads 4B-aligned

__device__ __forceinline__ unsigned short f2bf(float f) {
  unsigned int u = __float_as_uint(f);
  unsigned int r = (u + 0x7FFFu + ((u >> 16) & 1u)) >> 16;  // RNE
  return (unsigned short)r;
}
__device__ __forceinline__ float bf2f(unsigned short h) {
  return __uint_as_float(((unsigned int)h) << 16);
}

// ---------------- deterministic Frobenius norm (two-pass) ------------------

__device__ __forceinline__ float block_reduce_sum(float s) {
  __shared__ float red[8];
  int lane = threadIdx.x & 31, wave = threadIdx.x >> 5;
#pragma unroll
  for (int off = 16; off > 0; off >>= 1) s += __shfl_down(s, off, 32);
  if (lane == 0) red[wave] = s;
  __syncthreads();
  s = (threadIdx.x < 8) ? red[threadIdx.x] : 0.0f;
  if (wave == 0) {
#pragma unroll
    for (int off = 4; off > 0; off >>= 1) s += __shfl_down(s, off, 32);
  }
  return s;  // valid in thread 0
}

__global__ void k_sumsq(const float* __restrict__ X, float* __restrict__ partial) {
  const int b = blockIdx.y;
  const float* Xb = X + (size_t)b * MDIM * MDIM;
  int tid = blockIdx.x * 256 + threadIdx.x;
  float s = 0.0f;
  for (int i = tid; i < MDIM * MDIM; i += 256 * 256) {
    float v = Xb[i];
    s += v * v;
  }
  s = block_reduce_sum(s);
  if (threadIdx.x == 0) partial[b * 256 + blockIdx.x] = s;
}

__global__ void k_norm_final(const float* __restrict__ partial, float* __restrict__ norms) {
  float s = partial[blockIdx.x * 256 + threadIdx.x];
  s = block_reduce_sum(s);
  if (threadIdx.x == 0) norms[blockIdx.x] = s;
}

__global__ void k_scale(const float* __restrict__ X, const float* __restrict__ norms,
                        unsigned short* __restrict__ Xb) {
  size_t idx = (size_t)blockIdx.x * 256 + threadIdx.x;
  const size_t per = (size_t)MDIM * MDIM;
  if (idx >= (size_t)BATCH * per) return;
  int b = (int)(idx / per);
  float inv = INITIAL_SCALE * rsqrtf(norms[b]);
  Xb[idx] = f2bf(X[idx] * inv);
}

// ----------------------------- tiled WMMA GEMM -----------------------------
// C[b] = op(A[b]) * B[b],  op = A^T if TRANSA (S = X^T X) else A.
// POLY: Out = c0*Xc + c1*A + c2*acc (A aliases Y), optional fp32 copy.
// Block 256 thr = 8 waves; tile 128x128x32; wave tile 64x32 (4x2 WMMA tiles).
// Double-buffered LDS staging: global loads for tile i+1 are in flight while
// tile i's WMMAs issue; one barrier per k-step.

template <bool TRANSA, bool POLY>
__global__ __launch_bounds__(256) void k_gemm(
    const unsigned short* __restrict__ A,
    const unsigned short* __restrict__ Bm,
    unsigned short* __restrict__ Out,
    const unsigned short* __restrict__ Xc,
    const float* __restrict__ coef,
    float* __restrict__ OutF)
{
  __shared__ unsigned short sA[2][BM][BK + LDS_PAD];   // A tile, [m][k]
  __shared__ unsigned short sBt[2][BN][BK + LDS_PAD];  // B tile transposed [n][k]
  __shared__ float sC[8][16][16];                      // per-wave epilogue scratch

  const int b = blockIdx.z;
  const size_t matOff = (size_t)b * MDIM * MDIM;
  const unsigned short* Ab = A + matOff;
  const unsigned short* Bb = Bm + matOff;

  const int blockM = blockIdx.y * BM;
  const int blockN = blockIdx.x * BN;

  const int tid  = threadIdx.x;
  const int lane = tid & 31;
  const int wave = tid >> 5;
  const int wm = (wave >> 2) * 64;
  const int wn = (wave & 3) * 32;
  const int g   = lane >> 4;   // half-wave group (ISA 16-bit striping)
  const int l15 = lane & 15;

  v8f acc[4][2];
#pragma unroll
  for (int i = 0; i < 4; ++i)
#pragma unroll
    for (int j = 0; j < 2; ++j) acc[i][j] = (v8f){};

  union Frag { unsigned int d[8]; v16bf v; };

  // ---- staging helpers (one uint4 of 16 bf16 per thread per tile) ----
  auto loadA = [&](int kb) -> uint4 {
    if (TRANSA) {
      int k = tid >> 3, m0 = (tid & 7) * 16;
      return *(const uint4*)(Ab + (size_t)(kb + k) * MDIM + blockM + m0);
    } else {
      int m = tid >> 1, kh = (tid & 1) * 16;
      return *(const uint4*)(Ab + (size_t)(blockM + m) * MDIM + kb + kh);
    }
  };
  auto loadB = [&](int kb) -> uint4 {
    int k = tid >> 3, n0 = (tid & 7) * 16;
    return *(const uint4*)(Bb + (size_t)(kb + k) * MDIM + blockN + n0);
  };
  auto storeA = [&](int buf, uint4 q) {
    if (TRANSA) {
      union { uint4 q; unsigned short u[16]; } t; t.q = q;
      int k = tid >> 3, m0 = (tid & 7) * 16;
#pragma unroll
      for (int j = 0; j < 16; ++j) sA[buf][m0 + j][k] = t.u[j];
    } else {
      union { uint4 q; unsigned int u[8]; } t; t.q = q;
      int m = tid >> 1, kh = (tid & 1) * 16;
#pragma unroll
      for (int j = 0; j < 8; ++j) *(unsigned int*)&sA[buf][m][kh + 2 * j] = t.u[j];
    }
  };
  auto storeB = [&](int buf, uint4 q) {
    union { uint4 q; unsigned short u[16]; } t; t.q = q;
    int k = tid >> 3, n0 = (tid & 7) * 16;
#pragma unroll
    for (int j = 0; j < 16; ++j) sBt[buf][n0 + j][k] = t.u[j];
  };

  // ---- prologue: stage tile 0 ----
  storeA(0, loadA(0));
  storeB(0, loadB(0));
  __syncthreads();

  for (int i = 0; i < KSTEPS; ++i) {
    const int cur = i & 1;

    // issue global loads for tile i+1 (in flight during this tile's WMMAs)
    uint4 qa = make_uint4(0, 0, 0, 0), qb = make_uint4(0, 0, 0, 0);
    if (i + 1 < KSTEPS) {
      qa = loadA((i + 1) * BK);
      qb = loadB((i + 1) * BK);
    }
    if (i + 2 < KSTEPS) {  // global_prefetch_b8 for tile i+2
      int kb2 = (i + 2) * BK;
      if (TRANSA)
        __builtin_prefetch(Ab + (size_t)(kb2 + (tid >> 3)) * MDIM + blockM + (tid & 7) * 16, 0, 3);
      else
        __builtin_prefetch(Ab + (size_t)(blockM + (tid >> 1)) * MDIM + kb2 + (tid & 1) * 16, 0, 3);
      __builtin_prefetch(Bb + (size_t)(kb2 + (tid >> 3)) * MDIM + blockN + (tid & 7) * 16, 0, 3);
    }

    // ---- fragments (documented 16-bit WMMA VGPR striping) + 8 WMMAs ----
    Frag a[4], bf[2];
#pragma unroll
    for (int mi = 0; mi < 4; ++mi) {
      int row = wm + mi * 16 + l15;
#pragma unroll
      for (int v = 0; v < 8; ++v) {
        int k = ((v >= 4) ? 16 : 0) + g * 8 + (v & 3) * 2;
        a[mi].d[v] = *(const unsigned int*)&sA[cur][row][k];
      }
    }
#pragma unroll
    for (int ni = 0; ni < 2; ++ni) {
      int col = wn + ni * 16 + l15;
      int kbase = g * 8;
#pragma unroll
      for (int v = 0; v < 8; ++v)
        bf[ni].d[v] = *(const unsigned int*)&sBt[cur][col][((v >= 4) ? 16 : 0) + kbase + (v & 3) * 2];
    }
#pragma unroll
    for (int mi = 0; mi < 4; ++mi)
#pragma unroll
      for (int ni = 0; ni < 2; ++ni)
        acc[mi][ni] = __builtin_amdgcn_wmma_f32_16x16x32_bf16(
            false, a[mi].v, false, bf[ni].v, (short)0, acc[mi][ni], false, false);

    // ---- store next tile into the other buffer; one barrier per k-step ----
    if (i + 1 < KSTEPS) {
      storeA(1 - cur, qa);
      storeB(1 - cur, qb);
    }
    __syncthreads();
  }

  // ---- epilogue: transpose each 16x16 tile through per-wave LDS scratch so
  //      every lane owns 8 contiguous columns of a row -> b128 global I/O ----
  float c0 = 0.f, c1 = 0.f, c2 = 0.f;
  if (POLY) { c0 = coef[0]; c1 = coef[1]; c2 = coef[2]; }
  const int erow  = lane >> 1;       // 0..15
  const int ehalf = lane & 1;        // which 8-column half

#pragma unroll
  for (int mi = 0; mi < 4; ++mi) {
#pragma unroll
    for (int ni = 0; ni < 2; ++ni) {
      // scatter: lane owns column l15, rows g*8 .. g*8+7
#pragma unroll
      for (int r = 0; r < 8; ++r) sC[wave][g * 8 + r][l15] = acc[mi][ni][r];
      asm volatile("s_wait_dscnt 0x0" ::: "memory");  // wave-local LDS RAW fence

      float4 lo = *(const float4*)&sC[wave][erow][ehalf * 8];
      float4 hi = *(const float4*)&sC[wave][erow][ehalf * 8 + 4];
      float v[8] = {lo.x, lo.y, lo.z, lo.w, hi.x, hi.y, hi.z, hi.w};

      const int rowG = blockM + wm + mi * 16 + erow;
      const int colG = blockN + wn + ni * 16 + ehalf * 8;
      const size_t off = matOff + (size_t)rowG * MDIM + colG;

      if (POLY) {
        union { uint4 q; unsigned short u[8]; } xq, yq;
        xq.q = *(const uint4*)(Xc + off);   // current X (bf16)
        yq.q = *(const uint4*)(A + off);    // Y (bf16)
#pragma unroll
        for (int e = 0; e < 8; ++e)
          v[e] = c0 * bf2f(xq.u[e]) + c1 * bf2f(yq.u[e]) + c2 * v[e];
        if (OutF) {
          *(float4*)(OutF + off)     = make_float4(v[0], v[1], v[2], v[3]);
          *(float4*)(OutF + off + 4) = make_float4(v[4], v[5], v[6], v[7]);
        }
      }
      uint4 pk;
      pk.x = (unsigned)f2bf(v[0]) | ((unsigned)f2bf(v[1]) << 16);
      pk.y = (unsigned)f2bf(v[2]) | ((unsigned)f2bf(v[3]) << 16);
      pk.z = (unsigned)f2bf(v[4]) | ((unsigned)f2bf(v[5]) << 16);
      pk.w = (unsigned)f2bf(v[6]) | ((unsigned)f2bf(v[7]) << 16);
      *(uint4*)(Out + off) = pk;            // 8 bf16 = one b128 store
    }
  }
}

// ------------------------------- launcher ----------------------------------

extern "C" void kernel_launch(void* const* d_in, const int* in_sizes, int n_in,
                              void* d_out, int out_size, void* d_ws, size_t ws_size,
                              hipStream_t stream) {
  (void)in_sizes; (void)n_in; (void)out_size; (void)ws_size;
  const float* X    = (const float*)d_in[0];
  const float* coef = (const float*)d_in[1];
  float* out = (float*)d_out;

  char* ws = (char*)d_ws;
  float* partial = (float*)ws;                       // 64*256 f32 = 64 KB
  float* norms   = (float*)(ws + 65536);             // 64 f32
  unsigned short* Xb = (unsigned short*)(ws + 131072);
  const size_t matElems = (size_t)BATCH * MDIM * MDIM;
  unsigned short* S = Xb + matElems;                 // 128 MB each
  unsigned short* Y = S + matElems;

  k_sumsq<<<dim3(256, BATCH), 256, 0, stream>>>(X, partial);
  k_norm_final<<<BATCH, 256, 0, stream>>>(partial, norms);
  {
    size_t total = (size_t)BATCH * MDIM * MDIM;
    k_scale<<<(unsigned)((total + 255) / 256), 256, 0, stream>>>(X, norms, Xb);
  }

  dim3 gg(MDIM / BN, MDIM / BM, BATCH);
  for (int it = 0; it < NUM_ITER; ++it) {
    k_gemm<true,  false><<<gg, 256, 0, stream>>>(Xb, Xb, S, nullptr, nullptr, nullptr);
    k_gemm<false, false><<<gg, 256, 0, stream>>>(Xb, S, Y, nullptr, nullptr, nullptr);
    k_gemm<false, true ><<<gg, 256, 0, stream>>>(
        Y, S, Xb, Xb, coef, (it == NUM_ITER - 1) ? out : nullptr);
  }
}
